// MultiHeadAttention_36172214567817
// MI455X (gfx1250) — matrix operand order repaired and checked by reading, not compiled, not run on hardware
//
#include <hip/hip_runtime.h>

// MI455X (gfx1250) MHA forward. Compute-bound (~69 GFLOP): all GEMMs via
// v_wmma_f32_16x16x32_f16. Projections/out-proj use TDM (tensor_load_to_lds)
// double-buffered LDS staging with s_wait_tensorcnt; attention is flash-style
// online softmax with LDS C->A relayout of P.

typedef __attribute__((ext_vector_type(16))) _Float16 v16h;
typedef __attribute__((ext_vector_type(8)))  _Float16 v8h;
typedef __attribute__((ext_vector_type(8)))  float    v8f;
typedef __attribute__((ext_vector_type(4)))  unsigned int u32x4;
typedef __attribute__((ext_vector_type(8)))  int      i32x8;
typedef __attribute__((ext_vector_type(4)))  int      i32x4;

#define S_LEN 2048
#define EMB   1024
#define NHEAD 16
#define HDIM  64
#define BATCH 2
#define BHEAD (BATCH * NHEAD)   // 32
#define BSEQ  (BATCH * S_LEN)   // 4096

#define TM 128
#define TN 128
#define TK 32

__device__ __forceinline__ v16h cat8(v8h lo, v8h hi) {
  return __builtin_shufflevector(lo, hi, 0,1,2,3,4,5,6,7,8,9,10,11,12,13,14,15);
}

__device__ __forceinline__ v8f wmma16(v16h a, v16h b, v8f c) {
  return __builtin_amdgcn_wmma_f32_16x16x32_f16(false, a, false, b, (short)0, c,
                                                false, false);
}

__device__ __forceinline__ float redmax16(float v) {
  v = fmaxf(v, __shfl_xor(v, 1, 32));
  v = fmaxf(v, __shfl_xor(v, 2, 32));
  v = fmaxf(v, __shfl_xor(v, 4, 32));
  v = fmaxf(v, __shfl_xor(v, 8, 32));
  return v;
}

__device__ __forceinline__ float redsum16(float v) {
  v += __shfl_xor(v, 1, 32);
  v += __shfl_xor(v, 2, 32);
  v += __shfl_xor(v, 4, 32);
  v += __shfl_xor(v, 8, 32);
  return v;
}

// ---- TDM descriptor builders (CDNA5 ISA) ------
// 2D tile: tile_dim0 x tile_dim1 elements out of tensor (dim0 = contiguous).
__device__ __forceinline__ u32x4 tdm_g0(unsigned lds_addr, const void* gaddr) {
  unsigned long long ga = (unsigned long long)(uintptr_t)gaddr;
  u32x4 g;
  g.x = 1u;                                  // count=1 valid user descriptor
  g.y = lds_addr;                            // LDS byte address
  g.z = (unsigned)(ga & 0xFFFFFFFFu);        // global_addr[31:0]
  g.w = (unsigned)((ga >> 32) & 0x01FFFFFFu) // global_addr[56:32]
        | (2u << 30);                        // type = 2 ("image")
  return g;
}

__device__ __forceinline__ i32x8 tdm_g1(unsigned td0, unsigned td1,
                                        unsigned tile0, unsigned tile1,
                                        unsigned long long stride0) {
  i32x8 g;
  g[0] = (int)(1u << 16);                              // data_size = 2 bytes
  g[1] = (int)((td0 & 0xFFFFu) << 16);                 // tensor_dim0[15:0]
  g[2] = (int)((td0 >> 16) | ((td1 & 0xFFFFu) << 16)); // dim0 hi | dim1 lo
  g[3] = (int)((td1 >> 16) | (tile0 << 16));           // dim1 hi | tile_dim0
  g[4] = (int)(tile1 & 0xFFFFu);                       // tile_dim1 (tile_dim2=0)
  g[5] = (int)(stride0 & 0xFFFFFFFFull);               // dim0_stride[31:0]
  g[6] = (int)((stride0 >> 32) & 0xFFFFull);           // dim0_stride[47:32]
  g[7] = 0;
  return g;
}

__device__ __forceinline__ void tdm_load_2d(unsigned lds_addr, const void* gaddr,
                                            unsigned td0, unsigned td1,
                                            unsigned tile0, unsigned tile1,
                                            unsigned long long stride0) {
  i32x4 z4 = {0, 0, 0, 0};
  i32x8 z8 = {0, 0, 0, 0, 0, 0, 0, 0};
  // 6-arg form (clang-23 / therock-10.0): groups 2/3 + aux group unused (2D).
  __builtin_amdgcn_tensor_load_to_lds(tdm_g0(lds_addr, gaddr),
                                      tdm_g1(td0, td1, tile0, tile1, stride0),
                                      z4, z4, z8, 0);
}

// ---------------------------------------------------------------- casts
__global__ void cast_f32_to_f16(const float* __restrict__ in,
                                _Float16* __restrict__ out, int n) {
  int i = blockIdx.x * blockDim.x + threadIdx.x;
  if (i < n) out[i] = (_Float16)in[i];
}

// ---------------------------------------------------------------- GEMM
// Y[m,n] = sum_k X[m,k]*W[n,k] + bias[n].  Block tile 128x128, 8 waves in a
// 4x2 grid each computing 32x64. K streamed in 32-wide chunks by the TDM into
// double-buffered LDS. mode: 0 f16 [B,H,S,D]; 1 f16 [B,H,D,S]; 2 f32 [M,N].
__global__ __launch_bounds__(256)
void gemm_xwT(const _Float16* __restrict__ X, const _Float16* __restrict__ W,
              const float* __restrict__ bias, void* __restrict__ Y,
              int M, int N, int K, int mode) {
  __shared__ _Float16 sA[2][TM * TK];   // 8 KiB x2
  __shared__ _Float16 sB[2][TN * TK];   // 8 KiB x2
  const int lane = threadIdx.x & 31;
  const int lh   = lane & 15;
  const int sel  = lane >> 4;
  const int wv   = threadIdx.x >> 5;     // 0..7
  const int mt   = wv >> 1;              // 0..3 -> 32-row strip
  const int nt   = wv & 1;               // 0..1 -> 64-col strip
  const int tiles_n = N / TN;
  const int bm = (blockIdx.x / tiles_n) * TM;
  const int bn = (blockIdx.x % tiles_n) * TN;
  const bool issuer = (wv == 0);

  // prologue: DMA chunk 0 into buffer 0
  if (issuer) {
    tdm_load_2d((unsigned)(uintptr_t)&sA[0][0], X + (size_t)bm * K,
                (unsigned)K, (unsigned)M, TK, TM, (unsigned long long)K);
    tdm_load_2d((unsigned)(uintptr_t)&sB[0][0], W + (size_t)bn * K,
                (unsigned)K, (unsigned)N, TK, TN, (unsigned long long)K);
    __builtin_amdgcn_s_wait_tensorcnt(0);
  }
  __syncthreads();

  v8f acc[2][4] = {};
  int cur = 0;
  for (int k0 = 0; k0 < K; k0 += TK) {
    const int nxt = cur ^ 1;
    const bool more = (k0 + TK) < K;
    if (issuer && more) {   // overlap next chunk's DMA with this chunk's math
      tdm_load_2d((unsigned)(uintptr_t)&sA[nxt][0],
                  X + (size_t)bm * K + (k0 + TK),
                  (unsigned)K, (unsigned)M, TK, TM, (unsigned long long)K);
      tdm_load_2d((unsigned)(uintptr_t)&sB[nxt][0],
                  W + (size_t)bn * K + (k0 + TK),
                  (unsigned)K, (unsigned)N, TK, TN, (unsigned long long)K);
    }
    // A fragments: lane = row mt*32+i*16+lh; K = sel*16..+15 (contiguous)
    v16h a[2];
#pragma unroll
    for (int i = 0; i < 2; ++i) {
      const _Float16* p = &sA[cur][(mt * 32 + i * 16 + lh) * TK + sel * 16];
      a[i] = cat8(*(const v8h*)p, *(const v8h*)(p + 8));
    }
#pragma unroll
    for (int t = 0; t < 4; ++t) {
      // B fragment: lane = column nt*64+16t+lh; K = sel*16..+15 (contiguous)
      const _Float16* p = &sB[cur][(nt * 64 + 16 * t + lh) * TK + sel * 16];
      v16h b = cat8(*(const v8h*)p, *(const v8h*)(p + 8));
      acc[0][t] = wmma16(a[0], b, acc[0][t]);
      acc[1][t] = wmma16(a[1], b, acc[1][t]);
    }
    if (issuer && more) __builtin_amdgcn_s_wait_tensorcnt(0);
    __syncthreads();     // next buffer ready; this buffer free for reuse
    cur = nxt;
  }

#pragma unroll
  for (int i = 0; i < 2; ++i) {
#pragma unroll
    for (int t = 0; t < 4; ++t) {
      const int n = bn + nt * 64 + 16 * t + lh;
      const float bnv = bias[n];
#pragma unroll
      for (int r = 0; r < 8; ++r) {
        const float val = acc[i][t][r] + bnv;
        const int m = bm + mt * 32 + i * 16 + r + 8 * sel;
        if (mode == 2) {
          ((float*)Y)[(size_t)m * N + n] = val;
        } else {
          const int b = m >> 11, s = m & (S_LEN - 1);
          const int h = n >> 6,  d = n & (HDIM - 1);
          const _Float16 hv = (_Float16)val;
          if (mode == 0)
            ((_Float16*)Y)[(((size_t)(b * NHEAD + h) * S_LEN) + s) * HDIM + d] = hv;
          else
            ((_Float16*)Y)[(((size_t)(b * NHEAD + h) * HDIM) + d) * S_LEN + s] = hv;
        }
      }
    }
  }
}

// ---------------------------------------------------------------- attention
// One wave per (b,h, 16-query block). Causal flash attention, exp2 domain.
// Qh/Kh: [BH,S,D] f16, Vt: [BH,D,S] f16, Oh: [B,S,E] f16.
__global__ __launch_bounds__(128)
void flash_attn(const _Float16* __restrict__ Qh, const _Float16* __restrict__ Kh,
                const _Float16* __restrict__ Vt, _Float16* __restrict__ Oh) {
  __shared__ _Float16 pbuf[4][16 * 32];
  const int lane = threadIdx.x & 31;
  const int lh   = lane & 15;
  const int sel  = lane >> 4;
  const int wv   = threadIdx.x >> 5;
  const int wave = blockIdx.x * 4 + wv;
  const int bh   = wave >> 7;
  const int q0   = (wave & 127) << 4;
  _Float16* __restrict__ pds = &pbuf[wv][0];

  v16h aq[2];
#pragma unroll
  for (int c = 0; c < 2; ++c) {
    const _Float16* p = Qh + ((size_t)bh * S_LEN + q0 + lh) * HDIM + c * 32 + sel * 16;
    aq[c] = cat8(*(const v8h*)p, *(const v8h*)(p + 8));
  }

  float mrow[8], lrow[8], arow[8];
  v8f acc[4] = {};
#pragma unroll
  for (int r = 0; r < 8; ++r) { mrow[r] = -1e30f; lrow[r] = 0.f; }

  const float kscale = 0.18033688011f;   // log2(e)/sqrt(HDIM)
  const int jb_end = (q0 + 15) >> 5;
  for (int jb = 0; jb <= jb_end; ++jb) {
    const int j0 = jb << 5;
    v8f sc[2] = {};
#pragma unroll
    for (int c = 0; c < 2; ++c) {
#pragma unroll
      for (int t = 0; t < 2; ++t) {
        const _Float16* p = Kh + ((size_t)bh * S_LEN + j0 + 16 * t + lh) * HDIM
                               + c * 32 + sel * 16;
        v16h bk = cat8(*(const v8h*)p, *(const v8h*)(p + 8));
        sc[t] = wmma16(aq[c], bk, sc[t]);
      }
    }
#pragma unroll
    for (int r = 0; r < 8; ++r) {
      const int row = q0 + r + 8 * sel;
      const float t0 = (j0 + lh      <= row) ? sc[0][r] * kscale : -1e30f;
      const float t1 = (j0 + 16 + lh <= row) ? sc[1][r] * kscale : -1e30f;
      const float mnew = fmaxf(mrow[r], redmax16(fmaxf(t0, t1)));
      const float alpha = exp2f(mrow[r] - mnew);
      const float p0 = exp2f(t0 - mnew);
      const float p1 = exp2f(t1 - mnew);
      lrow[r] = lrow[r] * alpha + redsum16(p0 + p1);
      mrow[r] = mnew;
      arow[r] = alpha;
      pds[(r + 8 * sel) * 32 + lh]      = (_Float16)p0;
      pds[(r + 8 * sel) * 32 + 16 + lh] = (_Float16)p1;
    }
#pragma unroll
    for (int t = 0; t < 4; ++t)
#pragma unroll
      for (int r = 0; r < 8; ++r) acc[t][r] *= arow[r];

    const _Float16* pl = pds + lh * 32 + sel * 16;
    v16h ap = cat8(*(const v8h*)pl, *(const v8h*)(pl + 8));
#pragma unroll
    for (int t = 0; t < 4; ++t) {
      const _Float16* p = Vt + ((size_t)bh * HDIM + 16 * t + lh) * S_LEN
                             + j0 + sel * 16;
      v16h bv = cat8(*(const v8h*)p, *(const v8h*)(p + 8));
      acc[t] = wmma16(ap, bv, acc[t]);
    }
  }

  const int b = bh >> 4, h = bh & 15;
#pragma unroll
  for (int t = 0; t < 4; ++t) {
#pragma unroll
    for (int r = 0; r < 8; ++r) {
      const int row = q0 + r + 8 * sel;
      const float val = acc[t][r] / lrow[r];
      Oh[((size_t)(b * S_LEN) + row) * EMB + h * HDIM + 16 * t + lh] = (_Float16)val;
    }
  }
}

// ---------------------------------------------------------------- launch
extern "C" void kernel_launch(void* const* d_in, const int* in_sizes, int n_in,
                              void* d_out, int out_size, void* d_ws, size_t ws_size,
                              hipStream_t stream) {
  const float* q  = (const float*)d_in[0];
  const float* k  = (const float*)d_in[1];
  const float* v  = (const float*)d_in[2];
  // d_in[3]: causal tril mask -> implemented analytically in flash_attn
  const float* Wq = (const float*)d_in[4];
  const float* bq = (const float*)d_in[5];
  const float* Wk = (const float*)d_in[6];
  const float* bk = (const float*)d_in[7];
  const float* Wv = (const float*)d_in[8];
  const float* bv = (const float*)d_in[9];
  const float* Wo = (const float*)d_in[10];
  const float* bo = (const float*)d_in[11];

  char* w = (char*)d_ws;
  const size_t szX = (size_t)BSEQ * EMB * sizeof(_Float16);
  const size_t szW = (size_t)EMB * EMB * sizeof(_Float16);
  const size_t szQ = (size_t)BHEAD * S_LEN * HDIM * sizeof(_Float16);
  _Float16* q16  = (_Float16*)(w);
  _Float16* k16  = (_Float16*)(w + szX);
  _Float16* v16_ = (_Float16*)(w + 2 * szX);
  _Float16* Wq16 = (_Float16*)(w + 3 * szX);
  _Float16* Wk16 = (_Float16*)(w + 3 * szX + szW);
  _Float16* Wv16 = (_Float16*)(w + 3 * szX + 2 * szW);
  _Float16* Wo16 = (_Float16*)(w + 3 * szX + 3 * szW);
  _Float16* Qh   = (_Float16*)(w + 3 * szX + 4 * szW);
  _Float16* Kh   = (_Float16*)(w + 3 * szX + 4 * szW + szQ);
  _Float16* Vt   = (_Float16*)(w + 3 * szX + 4 * szW + 2 * szQ);
  _Float16* Oh   = (_Float16*)(w + 3 * szX + 4 * szW + 3 * szQ);

  const int nX = BSEQ * EMB, nW = EMB * EMB;
  cast_f32_to_f16<<<(nX + 255) / 256, 256, 0, stream>>>(q,  q16,  nX);
  cast_f32_to_f16<<<(nX + 255) / 256, 256, 0, stream>>>(k,  k16,  nX);
  cast_f32_to_f16<<<(nX + 255) / 256, 256, 0, stream>>>(v,  v16_, nX);
  cast_f32_to_f16<<<(nW + 255) / 256, 256, 0, stream>>>(Wq, Wq16, nW);
  cast_f32_to_f16<<<(nW + 255) / 256, 256, 0, stream>>>(Wk, Wk16, nW);
  cast_f32_to_f16<<<(nW + 255) / 256, 256, 0, stream>>>(Wv, Wv16, nW);
  cast_f32_to_f16<<<(nW + 255) / 256, 256, 0, stream>>>(Wo, Wo16, nW);

  // (BSEQ/128)*(EMB/128) = 256 blocks x 256 threads
  const dim3 gblk(256);
  const dim3 ggrd((BSEQ / TM) * (EMB / TN));
  gemm_xwT<<<ggrd, gblk, 0, stream>>>(q16,  Wq16, bq, Qh,   BSEQ, EMB, EMB, 0);
  gemm_xwT<<<ggrd, gblk, 0, stream>>>(k16,  Wk16, bk, Kh,   BSEQ, EMB, EMB, 0);
  gemm_xwT<<<ggrd, gblk, 0, stream>>>(v16_, Wv16, bv, Vt,   BSEQ, EMB, EMB, 1);

  flash_attn<<<dim3(BHEAD * (S_LEN / 16) / 4), dim3(128), 0, stream>>>(Qh, Kh, Vt, Oh);

  gemm_xwT<<<ggrd, gblk, 0, stream>>>(Oh, Wo16, bo, d_out, BSEQ, EMB, EMB, 2);
}
